// SemanGraphPPI_22874995818774
// MI455X (gfx1250) — matrix operand order repaired and checked by reading, not compiled, not run on hardware
//
#include <hip/hip_runtime.h>
#include <stdint.h>

typedef __bf16 bf16;
typedef __attribute__((ext_vector_type(16))) bf16  v16bf;
typedef __attribute__((ext_vector_type(8)))  bf16  v8bf;
typedef __attribute__((ext_vector_type(8)))  float v8f;

#define DIM     64
#define RREL    5
#define NW      6        // RREL relation mats + 1 root mat, fused per layer
#define LAYERS  10
#define N_MF    50000
#define E_MF    400000
#define N_BP    100000
#define E_BP    800000
#define N_CC    50000
#define E_CC    400000
#define NVOCAB  200000
#define NANN    400000
#define NPROT   20000
#define EPPI    400000
#define BPAIR   8192
#define NC      10
#define NHEAD   8

// ---------------------------------------------------------------------------
// bf16 WMMA GEMM, compile-time K:  C[M x N] = act(A[M x K] @ B[K x N] + bias)
//   - B pre-packed column-major: Bt[n*K + k] -> two 16-byte runs 16 elements apart per lane
//     per K-chunk (global_load_b128 pairs).
//   - A strip (16 rows x K, contiguous in global) async-copied to LDS ONCE per
//     block (ASYNCcnt), then reused for every 64-col chunk of N.
//   - Fully unrolled K loop: back-to-back v_wmma_f32_16x16x32_bf16 fed by
//     ds_load_b128.
// Requirements: K in {64,128,512}, N % 64 == 0. M arbitrary (guarded).
// Block = 128 threads (4 waves); grid.x covers M/16; N tiled in-kernel.
// ---------------------------------------------------------------------------
template <int KT>
__global__ void gemm_wmma_kernel(const bf16* __restrict__ A, const bf16* __restrict__ Bt,
                                 const float* __restrict__ bias, float* __restrict__ C,
                                 int M, int N, int relu)
{
    const int tid  = threadIdx.x;
    const int lane = tid & 31;
    const int wave = tid >> 5;
    const int m0   = blockIdx.x * 16;

    __shared__ __align__(16) bf16 As[16 * KT];

    const bool fullTile = (m0 + 16 <= M);
    if (fullTile) {
        // 16 x KT strip is one contiguous global range starting at A + m0*KT.
        const char* gbase   = (const char*)(A + (size_t)m0 * KT);
        unsigned    ldsbase = (unsigned)(uintptr_t)As;
#pragma unroll
        for (int c = tid * 16; c < 16 * KT * 2; c += 128 * 16) {
            unsigned           ldsoff = ldsbase + c;
            unsigned long long ga     = (unsigned long long)(uintptr_t)(gbase + c);
            asm volatile("global_load_async_to_lds_b128 %0, %1, off"
                         :: "v"(ldsoff), "v"(ga) : "memory");
        }
        asm volatile("s_wait_asynccnt 0" ::: "memory");
    } else {
        for (int i = tid * 4; i < 16 * KT; i += 128 * 4) {
            int row = i / KT;
            int gr  = m0 + row;
            if (gr < M) {
                const bf16* s = A + (size_t)gr * KT + (i % KT);
                As[i] = s[0]; As[i + 1] = s[1]; As[i + 2] = s[2]; As[i + 3] = s[3];
            } else {
                bf16 z = (bf16)0.0f;
                As[i] = z; As[i + 1] = z; As[i + 2] = z; As[i + 3] = z;
            }
        }
    }
    __syncthreads();

    // Per-lane fragment addressing (ISA 16-bit A 16x32 / B 32x16 layouts)
    const int mrow  = lane & 15;
    const int koff  = (lane >= 16) ? 8 : 0;    // A: K sub-offset
    const int kboff = (lane >= 16) ? 8 : 0;    // B: K sub-offset
    const int rbase = m0 + ((lane >= 16) ? 8 : 0);
    const int nIter = N >> 6;

    for (int it = 0; it < nIter; ++it) {
        const int   col  = it * 64 + wave * 16 + (lane & 15);
        const bf16* brow = Bt + (size_t)col * KT;
        __builtin_prefetch((const void*)brow, 0, 0);

        v8f acc = {};
#pragma unroll
        for (int k0 = 0; k0 < KT; k0 += 32) {
            // A: As[mrow][k0+koff .. +7] and As[mrow][k0+koff+16 .. +23]
            const v8bf* ap  = (const v8bf*)&As[mrow * KT + k0 + koff];
            v8bf        alo = ap[0];
            v8bf        ahi = ap[2];     // +16 elements
            v16bf a = __builtin_shufflevector(alo, ahi, 0, 1, 2, 3, 4, 5, 6, 7,
                                              8, 9, 10, 11, 12, 13, 14, 15);
            // B: Bt[col][k0+kboff .. +7] and Bt[col][k0+kboff+16 .. +23]
            const v8bf* bp  = (const v8bf*)(brow + k0 + kboff);
            v8bf        blo = bp[0];
            v8bf        bhi = bp[2];     // +16 elements
            v16bf b = __builtin_shufflevector(blo, bhi, 0, 1, 2, 3, 4, 5, 6, 7,
                                              8, 9, 10, 11, 12, 13, 14, 15);
            acc = __builtin_amdgcn_wmma_f32_16x16x32_bf16(false, a, false, b,
                                                          (short)0, acc, false, false);
        }

        // C layout: VGPR i -> row rbase + i, col fixed per lane
        const float bv = bias ? bias[col] : 0.0f;
        if (fullTile) {
#pragma unroll
            for (int i = 0; i < 8; ++i) {
                float v = acc[i] + bv;
                if (relu) v = fmaxf(v, 0.0f);
                C[(size_t)(rbase + i) * N + col] = v;
            }
        } else {
#pragma unroll
            for (int i = 0; i < 8; ++i) {
                int r = rbase + i;
                if (r < M) {
                    float v = acc[i] + bv;
                    if (relu) v = fmaxf(v, 0.0f);
                    C[(size_t)r * N + col] = v;
                }
            }
        }
    }
}

// ---------------------------------------------------------------------------
// Elementwise / utility kernels
// ---------------------------------------------------------------------------
__global__ void cvt_bf16_kernel(bf16* __restrict__ dst, const float* __restrict__ src, size_t n)
{
    size_t i = (size_t)blockIdx.x * blockDim.x + threadIdx.x;
    if (i < n) dst[i] = (bf16)src[i];
}

// Pack row-major W[k*N+n] (per matrix) into column-major bf16 Bt[n*K+k].
__global__ void pack_b_kernel(bf16* __restrict__ dst, const float* __restrict__ src,
                              int K, int N, int nmat)
{
    size_t idx = (size_t)blockIdx.x * blockDim.x + threadIdx.x;
    size_t per = (size_t)K * N;
    if (idx >= per * nmat) return;
    size_t mat = idx / per;
    size_t rem = idx % per;
    int n = (int)(rem / K), k = (int)(rem % K);
    dst[idx] = (bf16)src[mat * per + (size_t)k * N + n];
}

__global__ void fill_f32_kernel(float* p, float v, size_t n)
{
    size_t i = (size_t)blockIdx.x * blockDim.x + threadIdx.x;
    if (i < n) p[i] = v;
}
__global__ void fill_u32_kernel(unsigned* p, unsigned v, size_t n)
{
    size_t i = (size_t)blockIdx.x * blockDim.x + threadIdx.x;
    if (i < n) p[i] = v;
}

// ---------------------------------------------------------------------------
// Encoder (R-GCN) support.  xw layout: [node][NW*64], root slot = NW-1.
// ---------------------------------------------------------------------------
__global__ void edge_count_kernel(const int* __restrict__ dst, const int* __restrict__ et,
                                  float* __restrict__ cnt, int E)
{
    int e = blockIdx.x * blockDim.x + threadIdx.x;
    if (e < E) atomicAdd(&cnt[(size_t)dst[e] * RREL + et[e]], 1.0f);
}

__global__ void edge_agg_kernel(const int* __restrict__ src, const int* __restrict__ dst,
                                const int* __restrict__ et, const float* __restrict__ xw,
                                const float* __restrict__ cnt, float* __restrict__ agg,
                                int E)
{
    size_t idx = (size_t)blockIdx.x * blockDim.x + threadIdx.x;
    if (idx >= (size_t)E * 64) return;
    int e = (int)(idx >> 6), d = (int)(idx & 63);
    int s = src[e], t = dst[e], r = et[e];
    float c = fmaxf(cnt[(size_t)t * RREL + r], 1.0f);
    float v = xw[(size_t)s * (NW * 64) + r * 64 + d] / c;
    atomicAdd(&agg[(size_t)t * 64 + d], v);
}

// h_out = LN( relu(agg + root + bias) + h_in ) * g + b   (one wave per node)
// root rows have stride NW*64 (root slot inside fused xw).
__global__ void combine_ln_kernel(const float* __restrict__ hin, const float* __restrict__ agg,
                                  const float* __restrict__ root, const float* __restrict__ bias,
                                  const float* __restrict__ g, const float* __restrict__ bb,
                                  float* __restrict__ hout, int N)
{
    int lane = threadIdx.x & 31;
    int node = blockIdx.x * 8 + (threadIdx.x >> 5);
    if (node >= N) return;
    size_t base  = (size_t)node * 64;
    size_t rbase = (size_t)node * (NW * 64);
    int d1 = lane, d2 = lane + 32;
    float a1 = fmaxf(agg[base + d1] + root[rbase + d1] + bias[d1], 0.0f);
    float a2 = fmaxf(agg[base + d2] + root[rbase + d2] + bias[d2], 0.0f);
    float t1 = a1 + hin[base + d1];
    float t2 = a2 + hin[base + d2];
    float s = t1 + t2;
    for (int m = 16; m > 0; m >>= 1) s += __shfl_xor(s, m, 32);
    float mu = s * (1.0f / 64.0f);
    float c1 = t1 - mu, c2 = t2 - mu;
    float vs = c1 * c1 + c2 * c2;
    for (int m = 16; m > 0; m >>= 1) vs += __shfl_xor(vs, m, 32);
    float rs = rsqrtf(vs * (1.0f / 64.0f) + 1e-5f);
    hout[base + d1] = c1 * rs * g[d1] + bb[d1];
    hout[base + d2] = c2 * rs * g[d2] + bb[d2];
}

// ---------------------------------------------------------------------------
// Annotation pooling / segment softmax
// ---------------------------------------------------------------------------
__device__ __forceinline__ unsigned fenc(float f)
{
    unsigned u = __float_as_uint(f);
    return (u >> 31) ? ~u : (u | 0x80000000u);
}
__device__ __forceinline__ float fdec(unsigned u)
{
    return __uint_as_float((u & 0x80000000u) ? (u & 0x7FFFFFFFu) : ~u);
}

__global__ void ann_score_kernel(const float* __restrict__ voca, const int* __restrict__ annIdx,
                                 const float* __restrict__ poolW, const float* __restrict__ poolB,
                                 float* __restrict__ score, int n)
{
    int lane = threadIdx.x & 31;
    int i = blockIdx.x * 8 + (threadIdx.x >> 5);
    if (i >= n) return;
    size_t base = (size_t)annIdx[i] * 64;
    float s = voca[base + lane] * poolW[lane] + voca[base + lane + 32] * poolW[lane + 32];
    for (int m = 16; m > 0; m >>= 1) s += __shfl_xor(s, m, 32);
    if (lane == 0) score[i] = s + poolB[0];
}

__global__ void seg_max_kernel(const float* __restrict__ s, const int* __restrict__ seg,
                               unsigned* __restrict__ menc, int n)
{
    int i = blockIdx.x * blockDim.x + threadIdx.x;
    if (i < n) atomicMax(&menc[seg[i]], fenc(s[i]));
}

__global__ void seg_expsum_kernel(const float* __restrict__ s, const int* __restrict__ seg,
                                  const unsigned* __restrict__ menc,
                                  float* __restrict__ e, float* __restrict__ z, int n)
{
    int i = blockIdx.x * blockDim.x + threadIdx.x;
    if (i >= n) return;
    int g = seg[i];
    float ev = __expf(s[i] - fdec(menc[g]));
    e[i] = ev;
    atomicAdd(&z[g], ev);
}

__global__ void pool_accum_kernel(const float* __restrict__ e, const float* __restrict__ z,
                                  const int* __restrict__ seg, const int* __restrict__ annIdx,
                                  const float* __restrict__ voca, float* __restrict__ prot, int n)
{
    size_t idx = (size_t)blockIdx.x * blockDim.x + threadIdx.x;
    if (idx >= (size_t)n * 64) return;
    int i = (int)(idx >> 6), d = (int)(idx & 63);
    int g = seg[i];
    float a = e[i] / fmaxf(z[g], 1e-16f);
    atomicAdd(&prot[(size_t)g * 64 + d], a * voca[(size_t)annIdx[i] * 64 + d]);
}

// ---------------------------------------------------------------------------
// GCN support
// ---------------------------------------------------------------------------
__global__ void deg_kernel(const int* __restrict__ dst, float* __restrict__ deg, int E)
{
    int e = blockIdx.x * blockDim.x + threadIdx.x;
    if (e < E) atomicAdd(&deg[dst[e]], 1.0f);
}
__global__ void dinv_kernel(const float* __restrict__ deg, float* __restrict__ dinv, int n)
{
    int i = blockIdx.x * blockDim.x + threadIdx.x;
    if (i < n) dinv[i] = rsqrtf(fmaxf(deg[i] + 1.0f, 1.0f)); // +1 self loop
}
__global__ void gcn_edge_kernel(const int* __restrict__ src, const int* __restrict__ dst,
                                const float* __restrict__ h, const float* __restrict__ dinv,
                                float* __restrict__ out, int E)
{
    size_t idx = (size_t)blockIdx.x * blockDim.x + threadIdx.x;
    if (idx >= (size_t)E * 64) return;
    int e = (int)(idx >> 6), d = (int)(idx & 63);
    int s = src[e], t = dst[e];
    atomicAdd(&out[(size_t)t * 64 + d], h[(size_t)s * 64 + d] * dinv[s] * dinv[t]);
}
__global__ void gcn_self_bias_kernel(const float* __restrict__ h, const float* __restrict__ dinv,
                                     const float* __restrict__ bias, float* __restrict__ out, int n)
{
    size_t idx = (size_t)blockIdx.x * blockDim.x + threadIdx.x;
    if (idx >= (size_t)n * 64) return;
    int nn = (int)(idx >> 6), d = (int)(idx & 63);
    out[idx] += h[idx] * dinv[nn] * dinv[nn] + bias[d];
}

// ---------------------------------------------------------------------------
// MHA: one block per (query l, head h); online softmax over 20000 keys
// ---------------------------------------------------------------------------
__global__ void mha_attn_kernel(const float* __restrict__ Q, const float* __restrict__ Kb,
                                const float* __restrict__ Vb, float* __restrict__ O, int nk)
{
    __shared__ float sm[256], sz[256], sv[256 * 8];
    int tid = threadIdx.x;
    int l = blockIdx.x >> 3;
    int h = blockIdx.x & 7;
    float q[8];
#pragma unroll
    for (int j = 0; j < 8; ++j) q[j] = Q[l * 64 + h * 8 + j];
    const float scale = 0.35355339059327373f; // 1/sqrt(8)
    float m = -1e30f, z = 0.f, v[8] = {0, 0, 0, 0, 0, 0, 0, 0};
    for (int s = tid; s < nk; s += 256) {
        const float* kr = Kb + (size_t)s * 64 + h * 8;
        float sc = 0.f;
#pragma unroll
        for (int j = 0; j < 8; ++j) sc += q[j] * kr[j];
        sc *= scale;
        float nm = fmaxf(m, sc);
        float r = __expf(m - nm), w = __expf(sc - nm);
        z = z * r + w;
        const float* vr = Vb + (size_t)s * 64 + h * 8;
#pragma unroll
        for (int j = 0; j < 8; ++j) v[j] = v[j] * r + w * vr[j];
        m = nm;
    }
    sm[tid] = m; sz[tid] = z;
#pragma unroll
    for (int j = 0; j < 8; ++j) sv[tid * 8 + j] = v[j];
    __syncthreads();
    for (int off = 128; off > 0; off >>= 1) {
        if (tid < off) {
            float m1 = sm[tid], m2 = sm[tid + off];
            float nm = fmaxf(m1, m2);
            float r1 = __expf(m1 - nm), r2 = __expf(m2 - nm);
            sz[tid] = sz[tid] * r1 + sz[tid + off] * r2;
            for (int j = 0; j < 8; ++j)
                sv[tid * 8 + j] = sv[tid * 8 + j] * r1 + sv[(tid + off) * 8 + j] * r2;
            sm[tid] = nm;
        }
        __syncthreads();
    }
    if (tid < 8) O[l * 64 + h * 8 + tid] = sv[tid] / fmaxf(sz[0], 1e-30f);
}

__global__ void context_kernel(const float* __restrict__ cl, float* __restrict__ ctx)
{
    int d = threadIdx.x;
    if (d < 64) {
        float s = 0.f;
        for (int l = 0; l < NC; ++l) s += cl[l * 64 + d];
        ctx[d] = s * (1.0f / NC);
    }
}

// ---------------------------------------------------------------------------
// GAT support
// ---------------------------------------------------------------------------
__global__ void gat_node_kernel(const float* __restrict__ h, const float* __restrict__ asrc,
                                const float* __restrict__ adst, float* __restrict__ hs,
                                float* __restrict__ hd, int n)
{
    int lane = threadIdx.x & 31;
    int i = blockIdx.x * 8 + (threadIdx.x >> 5);
    if (i >= n) return;
    size_t b = (size_t)i * 64;
    float s1 = h[b + lane] * asrc[lane] + h[b + lane + 32] * asrc[lane + 32];
    float s2 = h[b + lane] * adst[lane] + h[b + lane + 32] * adst[lane + 32];
    for (int m = 16; m > 0; m >>= 1) { s1 += __shfl_xor(s1, m, 32); s2 += __shfl_xor(s2, m, 32); }
    if (lane == 0) { hs[i] = s1; hd[i] = s2; }
}

__global__ void dot64_kernel(const float* __restrict__ a, const float* __restrict__ b,
                             float* __restrict__ out)
{
    int lane = threadIdx.x;
    float s = a[lane] * b[lane] + a[lane + 32] * b[lane + 32];
    for (int m = 16; m > 0; m >>= 1) s += __shfl_xor(s, m, 32);
    if (lane == 0) out[0] = s;
}

__global__ void gat_edge_kernel(const int* __restrict__ src, const int* __restrict__ dst,
                                const float* __restrict__ hs, const float* __restrict__ hd,
                                const float* __restrict__ cdot, float* __restrict__ e, int E)
{
    int i = blockIdx.x * blockDim.x + threadIdx.x;
    if (i >= E) return;
    float v = hs[src[i]] + hd[dst[i]] + cdot[0];
    e[i] = (v > 0.f) ? v : 0.2f * v;
}

__global__ void gat_accum_kernel(const float* __restrict__ e, const float* __restrict__ z,
                                 const int* __restrict__ src, const int* __restrict__ dst,
                                 const float* __restrict__ h, float* __restrict__ xg, int E)
{
    size_t idx = (size_t)blockIdx.x * blockDim.x + threadIdx.x;
    if (idx >= (size_t)E * 64) return;
    int i = (int)(idx >> 6), d = (int)(idx & 63);
    int t = dst[i];
    float a = e[i] / fmaxf(z[t], 1e-16f);
    atomicAdd(&xg[(size_t)t * 64 + d], a * h[(size_t)src[i] * 64 + d]);
}

__global__ void add_bias64_kernel(float* __restrict__ x, const float* __restrict__ b, int n)
{
    size_t idx = (size_t)blockIdx.x * blockDim.x + threadIdx.x;
    if (idx < (size_t)n * 64) x[idx] += b[idx & 63];
}

// ---------------------------------------------------------------------------
// Pair head
// ---------------------------------------------------------------------------
__global__ void pair_kernel(const float* __restrict__ xg, const int* __restrict__ p1,
                            const int* __restrict__ p2, float* __restrict__ hc, int nB)
{
    size_t idx = (size_t)blockIdx.x * blockDim.x + threadIdx.x;
    if (idx >= (size_t)nB * 128) return;
    int i = (int)(idx >> 7), j = (int)(idx & 127);
    hc[idx] = (j < 64) ? xg[(size_t)p1[i] * 64 + j] : xg[(size_t)p2[i] * 64 + (j - 64)];
}

__global__ void final_out_kernel(const float* __restrict__ f2, const float* __restrict__ W,
                                 const float* __restrict__ bO, float* __restrict__ out, int n)
{
    int lane = threadIdx.x & 31;
    int i = blockIdx.x * 8 + (threadIdx.x >> 5);
    if (i >= n) return;
    const float* r = f2 + (size_t)i * 256;
    float s = 0.f;
#pragma unroll
    for (int t = 0; t < 8; ++t) s += r[lane + 32 * t] * W[lane + 32 * t];
    for (int m = 16; m > 0; m >>= 1) s += __shfl_xor(s, m, 32);
    if (lane == 0) out[i] = s + bO[0];
}

// ---------------------------------------------------------------------------
// Host orchestration
// ---------------------------------------------------------------------------
extern "C" void kernel_launch(void* const* d_in, const int* in_sizes, int n_in,
                              void* d_out, int out_size, void* d_ws, size_t ws_size,
                              hipStream_t stream)
{
    (void)in_sizes; (void)n_in; (void)out_size; (void)ws_size;
    char* base = (char*)d_ws;
    size_t off = 0;
    auto alloc = [&](size_t bytes) -> char* {
        char* p = base + off;
        off += (bytes + 255) & ~(size_t)255;
        return p;
    };

    const int MAXN = N_BP;
    float* hA      = (float*)alloc((size_t)MAXN * 64 * 4);
    float* hB_     = (float*)alloc((size_t)MAXN * 64 * 4);
    bf16*  hbf     = (bf16*) alloc((size_t)MAXN * 64 * 2);
    bf16*  wallbf  = (bf16*) alloc((size_t)LAYERS * NW * 4096 * 2);  // fused per-layer weights
    float* cnt     = (float*)alloc((size_t)MAXN * RREL * 4);
    float* agg     = (float*)alloc((size_t)MAXN * 64 * 4);
    float* voca    = (float*)alloc((size_t)NVOCAB * 64 * 4);
    char*  pool    = alloc((size_t)NW * MAXN * 64 * 4);   // xw + phase-2 pool
    float* xw      = (float*)pool;

    auto fillf = [&](float* p, float v, size_t n) {
        fill_f32_kernel<<<dim3((unsigned)((n + 255) / 256)), 256, 0, stream>>>(p, v, n);
    };
    auto fillu = [&](unsigned* p, unsigned v, size_t n) {
        fill_u32_kernel<<<dim3((unsigned)((n + 255) / 256)), 256, 0, stream>>>(p, v, n);
    };
    auto cvt = [&](bf16* d, const float* s, size_t n) {
        cvt_bf16_kernel<<<dim3((unsigned)((n + 255) / 256)), 256, 0, stream>>>(d, s, n);
    };
    auto packB = [&](bf16* d, const float* s, int K, int N, int nmat) {
        size_t tot = (size_t)K * N * nmat;
        pack_b_kernel<<<dim3((unsigned)((tot + 255) / 256)), 256, 0, stream>>>(d, s, K, N, nmat);
    };
    auto gemm = [&](const bf16* A, const bf16* Bt, const float* bias, float* C,
                    int M, int N, int K, int relu) {
        dim3 g((unsigned)((M + 15) / 16));
        if (K == 64)
            gemm_wmma_kernel<64><<<g, 128, 0, stream>>>(A, Bt, bias, C, M, N, relu);
        else if (K == 128)
            gemm_wmma_kernel<128><<<g, 128, 0, stream>>>(A, Bt, bias, C, M, N, relu);
        else
            gemm_wmma_kernel<512><<<g, 128, 0, stream>>>(A, Bt, bias, C, M, N, relu);
    };

    // ---------------- Phase 1: three R-GCN encoders -> voca -----------------
    struct GDef { int ix; int N; int E; size_t voff; };
    const GDef gdefs[3] = {
        { 0,  N_MF, E_MF, (size_t)0 },
        { 8,  N_BP, E_BP, (size_t)N_MF * 64 },
        { 16, N_CC, E_CC, (size_t)(N_MF + N_BP) * 64 },
    };
    for (int gi = 0; gi < 3; ++gi) {
        const GDef& G = gdefs[gi];
        const float* x     = (const float*)d_in[G.ix + 0];
        const int*   edges = (const int*)  d_in[G.ix + 1];
        const int*   et    = (const int*)  d_in[G.ix + 2];
        const float* Wrel  = (const float*)d_in[G.ix + 3];
        const float* Wroot = (const float*)d_in[G.ix + 4];
        const float* biasL = (const float*)d_in[G.ix + 5];
        const float* lng   = (const float*)d_in[G.ix + 6];
        const float* lnb   = (const float*)d_in[G.ix + 7];
        const int N = G.N, E = G.E;
        const int* esrc = edges;
        const int* edst = edges + E;

        // Fused per-layer weight block: [5 relation mats | root mat], each
        // packed column-major -> one N=384 GEMM per layer.
        for (int l = 0; l < LAYERS; ++l) {
            packB(wallbf + (size_t)l * NW * 4096, Wrel + (size_t)l * RREL * 4096, 64, 64, RREL);
            packB(wallbf + (size_t)l * NW * 4096 + (size_t)RREL * 4096,
                  Wroot + (size_t)l * 4096, 64, 64, 1);
        }
        hipMemcpyAsync(hA, x, (size_t)N * 64 * 4, hipMemcpyDeviceToDevice, stream);

        float* hcur = hA;
        float* hnxt = hB_;
        for (int l = 0; l < LAYERS; ++l) {
            cvt(hbf, hcur, (size_t)N * 64);
            gemm(hbf, wallbf + (size_t)l * NW * 4096, nullptr, xw, N, NW * 64, 64, 0);
            fillf(cnt, 0.f, (size_t)N * RREL);
            fillf(agg, 0.f, (size_t)N * 64);
            edge_count_kernel<<<dim3((unsigned)((E + 255) / 256)), 256, 0, stream>>>(edst, et, cnt, E);
            edge_agg_kernel<<<dim3((unsigned)(((size_t)E * 64 + 255) / 256)), 256, 0, stream>>>(
                esrc, edst, et, xw, cnt, agg, E);
            combine_ln_kernel<<<dim3((unsigned)((N + 7) / 8)), 256, 0, stream>>>(
                hcur, agg, xw + (size_t)RREL * 64, biasL + l * 64, lng + l * 64, lnb + l * 64,
                hnxt, N);
            float* t = hcur; hcur = hnxt; hnxt = t;
        }
        hipMemcpyAsync(voca + G.voff, hcur, (size_t)N * 64 * 4, hipMemcpyDeviceToDevice, stream);
    }

    // ---------------- Phase 2: pooling, GCN, MHA, GAT, MLP ------------------
    const int*   annIdx   = (const int*)  d_in[24];
    const int*   annBatch = (const int*)  d_in[25];
    const int*   pid1     = (const int*)  d_in[26];
    const int*   pid2     = (const int*)  d_in[27];
    const int*   ppi      = (const int*)  d_in[28];
    const float* poolW    = (const float*)d_in[29];
    const float* poolB    = (const float*)d_in[30];
    const float* gKW      = (const float*)d_in[31];
    const float* gKb      = (const float*)d_in[32];
    const float* gVW      = (const float*)d_in[33];
    const float* gVb      = (const float*)d_in[34];
    const float* superQ   = (const float*)d_in[35];
    const float* mhaInW   = (const float*)d_in[36];
    const float* mhaInB   = (const float*)d_in[37];
    const float* mhaOutW  = (const float*)d_in[38];
    const float* mhaOutB  = (const float*)d_in[39];
    const float* gatW     = (const float*)d_in[40];
    const float* gatB     = (const float*)d_in[41];
    const float* gatAsrc  = (const float*)d_in[42];
    const float* gatAdst  = (const float*)d_in[43];
    const float* gatActx  = (const float*)d_in[44];
    const float* t1W      = (const float*)d_in[45];
    const float* t1b      = (const float*)d_in[46];
    const float* t2W      = (const float*)d_in[47];
    const float* t2b      = (const float*)d_in[48];
    const float* fc1W     = (const float*)d_in[49];
    const float* fc1b     = (const float*)d_in[50];
    const float* fc2W     = (const float*)d_in[51];
    const float* fc2b     = (const float*)d_in[52];
    const float* outW     = (const float*)d_in[53];
    const float* outB     = (const float*)d_in[54];
    const int* psrc = ppi;
    const int* pdst = ppi + EPPI;

    size_t poff = 0;
    auto palloc = [&](size_t bytes) -> char* {
        char* p = pool + poff;
        poff += (bytes + 255) & ~(size_t)255;
        return p;
    };
    float*    score  = (float*)   palloc((size_t)NANN * 4);
    float*    eAnn   = (float*)   palloc((size_t)NANN * 4);
    unsigned* menc   = (unsigned*)palloc((size_t)NPROT * 4);
    float*    zseg   = (float*)   palloc((size_t)NPROT * 4);
    float*    prot   = (float*)   palloc((size_t)NPROT * 64 * 4);
    bf16*     protbf = (bf16*)    palloc((size_t)NPROT * 64 * 2);
    float*    degv   = (float*)   palloc((size_t)NPROT * 4);
    float*    dinv   = (float*)   palloc((size_t)NPROT * 4);
    float*    hK     = (float*)   palloc((size_t)NPROT * 64 * 4);
    float*    hV     = (float*)   palloc((size_t)NPROT * 64 * 4);
    float*    xK     = (float*)   palloc((size_t)NPROT * 64 * 4);
    float*    xV     = (float*)   palloc((size_t)NPROT * 64 * 4);
    bf16*     xKbf   = (bf16*)    palloc((size_t)NPROT * 64 * 2);
    bf16*     xVbf   = (bf16*)    palloc((size_t)NPROT * 64 * 2);
    bf16*     gKbf   = (bf16*)    palloc(4096 * 2);
    bf16*     gVbf   = (bf16*)    palloc(4096 * 2);
    bf16*     WqT    = (bf16*)    palloc(4096 * 2);
    bf16*     WkT    = (bf16*)    palloc(4096 * 2);
    bf16*     WvT    = (bf16*)    palloc(4096 * 2);
    bf16*     moutT  = (bf16*)    palloc(4096 * 2);
    bf16*     gatbf  = (bf16*)    palloc(4096 * 2);
    bf16*     t1bf   = (bf16*)    palloc((size_t)64 * 128 * 2);
    bf16*     t2bf   = (bf16*)    palloc((size_t)128 * 64 * 2);
    bf16*     fc1bf  = (bf16*)    palloc((size_t)128 * 512 * 2);
    bf16*     fc2bf  = (bf16*)    palloc((size_t)512 * 256 * 2);
    bf16*     sqbf   = (bf16*)    palloc((size_t)NC * 64 * 2);
    float*    Qf     = (float*)   palloc((size_t)NC * 64 * 4);
    float*    Kf     = (float*)   palloc((size_t)NPROT * 64 * 4);
    float*    Vf     = (float*)   palloc((size_t)NPROT * 64 * 4);
    float*    oF     = (float*)   palloc((size_t)NC * 64 * 4);
    bf16*     obf    = (bf16*)    palloc((size_t)NC * 64 * 2);
    float*    clF    = (float*)   palloc((size_t)NC * 64 * 4);
    float*    ctx    = (float*)   palloc(64 * 4);
    float*    cdot   = (float*)   palloc(4);
    float*    hG     = (float*)   palloc((size_t)NPROT * 64 * 4);
    float*    hsv    = (float*)   palloc((size_t)NPROT * 4);
    float*    hdv    = (float*)   palloc((size_t)NPROT * 4);
    float*    eG     = (float*)   palloc((size_t)EPPI * 4);
    float*    eExp   = (float*)   palloc((size_t)EPPI * 4);
    float*    xg     = (float*)   palloc((size_t)NPROT * 64 * 4);
    bf16*     xgbf   = (bf16*)    palloc((size_t)NPROT * 64 * 2);
    float*    xg2    = (float*)   palloc((size_t)NPROT * 128 * 4);
    bf16*     xg2bf  = (bf16*)    palloc((size_t)NPROT * 128 * 2);
    float*    xg3    = (float*)   palloc((size_t)NPROT * 64 * 4);
    float*    hc     = (float*)   palloc((size_t)BPAIR * 128 * 4);
    bf16*     hcbf   = (bf16*)    palloc((size_t)BPAIR * 128 * 2);
    float*    f1     = (float*)   palloc((size_t)BPAIR * 512 * 4);
    bf16*     f1bf   = (bf16*)    palloc((size_t)BPAIR * 512 * 2);
    float*    f2     = (float*)   palloc((size_t)BPAIR * 256 * 4);

    // ---- annotation attention pooling -> prot ----
    ann_score_kernel<<<dim3((NANN + 7) / 8), 256, 0, stream>>>(voca, annIdx, poolW, poolB, score, NANN);
    fillu(menc, 0u, NPROT);
    fillf(zseg, 0.f, NPROT);
    fillf(prot, 0.f, (size_t)NPROT * 64);
    seg_max_kernel<<<dim3((NANN + 255) / 256), 256, 0, stream>>>(score, annBatch, menc, NANN);
    seg_expsum_kernel<<<dim3((NANN + 255) / 256), 256, 0, stream>>>(score, annBatch, menc, eAnn, zseg, NANN);
    pool_accum_kernel<<<dim3((unsigned)(((size_t)NANN * 64 + 255) / 256)), 256, 0, stream>>>(
        eAnn, zseg, annBatch, annIdx, voca, prot, NANN);

    // ---- two GCNs (K, V) over PPI graph ----
    fillf(degv, 0.f, NPROT);
    deg_kernel<<<dim3((EPPI + 255) / 256), 256, 0, stream>>>(pdst, degv, EPPI);
    dinv_kernel<<<dim3((NPROT + 255) / 256), 256, 0, stream>>>(degv, dinv, NPROT);
    cvt(protbf, prot, (size_t)NPROT * 64);
    packB(gKbf, gKW, 64, 64, 1);
    packB(gVbf, gVW, 64, 64, 1);
    gemm(protbf, gKbf, nullptr, hK, NPROT, 64, 64, 0);
    gemm(protbf, gVbf, nullptr, hV, NPROT, 64, 64, 0);
    fillf(xK, 0.f, (size_t)NPROT * 64);
    fillf(xV, 0.f, (size_t)NPROT * 64);
    gcn_edge_kernel<<<dim3((unsigned)(((size_t)EPPI * 64 + 255) / 256)), 256, 0, stream>>>(psrc, pdst, hK, dinv, xK, EPPI);
    gcn_edge_kernel<<<dim3((unsigned)(((size_t)EPPI * 64 + 255) / 256)), 256, 0, stream>>>(psrc, pdst, hV, dinv, xV, EPPI);
    gcn_self_bias_kernel<<<dim3((unsigned)(((size_t)NPROT * 64 + 255) / 256)), 256, 0, stream>>>(hK, dinv, gKb, xK, NPROT);
    gcn_self_bias_kernel<<<dim3((unsigned)(((size_t)NPROT * 64 + 255) / 256)), 256, 0, stream>>>(hV, dinv, gVb, xV, NPROT);

    // ---- MHA (super queries vs proteins) ----
    // q @ W.T means column-major pack == direct convert of the weight rows.
    cvt(WqT,   mhaInW,          4096);
    cvt(WkT,   mhaInW + 4096,   4096);
    cvt(WvT,   mhaInW + 8192,   4096);
    cvt(moutT, mhaOutW,         4096);
    cvt(sqbf, superQ, (size_t)NC * 64);
    cvt(xKbf, xK, (size_t)NPROT * 64);
    cvt(xVbf, xV, (size_t)NPROT * 64);
    gemm(sqbf, WqT, mhaInB,       Qf, NC,    64, 64, 0);
    gemm(xKbf, WkT, mhaInB + 64,  Kf, NPROT, 64, 64, 0);
    gemm(xVbf, WvT, mhaInB + 128, Vf, NPROT, 64, 64, 0);
    mha_attn_kernel<<<NC * NHEAD, 256, 0, stream>>>(Qf, Kf, Vf, oF, NPROT);
    cvt(obf, oF, (size_t)NC * 64);
    gemm(obf, moutT, mhaOutB, clF, NC, 64, 64, 0);
    context_kernel<<<1, 64, 0, stream>>>(clF, ctx);

    // ---- GAT over PPI graph with context ----
    packB(gatbf, gatW, 64, 64, 1);
    gemm(xVbf, gatbf, nullptr, hG, NPROT, 64, 64, 0);
    gat_node_kernel<<<dim3((NPROT + 7) / 8), 256, 0, stream>>>(hG, gatAsrc, gatAdst, hsv, hdv, NPROT);
    dot64_kernel<<<1, 32, 0, stream>>>(ctx, gatActx, cdot);
    gat_edge_kernel<<<dim3((EPPI + 255) / 256), 256, 0, stream>>>(psrc, pdst, hsv, hdv, cdot, eG, EPPI);
    fillu(menc, 0u, NPROT);
    fillf(zseg, 0.f, NPROT);
    seg_max_kernel<<<dim3((EPPI + 255) / 256), 256, 0, stream>>>(eG, pdst, menc, EPPI);
    seg_expsum_kernel<<<dim3((EPPI + 255) / 256), 256, 0, stream>>>(eG, pdst, menc, eExp, zseg, EPPI);
    fillf(xg, 0.f, (size_t)NPROT * 64);
    gat_accum_kernel<<<dim3((unsigned)(((size_t)EPPI * 64 + 255) / 256)), 256, 0, stream>>>(
        eExp, zseg, psrc, pdst, hG, xg, EPPI);
    add_bias64_kernel<<<dim3((unsigned)(((size_t)NPROT * 64 + 255) / 256)), 256, 0, stream>>>(xg, gatB, NPROT);

    // ---- node transforms ----
    packB(t1bf, t1W, 64, 128, 1);
    packB(t2bf, t2W, 128, 64, 1);
    cvt(xgbf, xg, (size_t)NPROT * 64);
    gemm(xgbf, t1bf, t1b, xg2, NPROT, 128, 64, 1);
    cvt(xg2bf, xg2, (size_t)NPROT * 128);
    gemm(xg2bf, t2bf, t2b, xg3, NPROT, 64, 128, 1);

    // ---- pair MLP head ----
    pair_kernel<<<dim3((unsigned)(((size_t)BPAIR * 128 + 255) / 256)), 256, 0, stream>>>(xg3, pid1, pid2, hc, BPAIR);
    cvt(hcbf, hc, (size_t)BPAIR * 128);
    packB(fc1bf, fc1W, 128, 512, 1);
    packB(fc2bf, fc2W, 512, 256, 1);
    gemm(hcbf, fc1bf, fc1b, f1, BPAIR, 512, 128, 1);
    cvt(f1bf, f1, (size_t)BPAIR * 512);
    gemm(f1bf, fc2bf, fc2b, f2, BPAIR, 256, 512, 1);
    final_out_kernel<<<dim3((BPAIR + 7) / 8), 256, 0, stream>>>(f2, outW, outB, (float*)d_out, BPAIR);
}
